// HGT_OL_74509092651089
// MI455X (gfx1250) — matrix operand order repaired and checked by reading, not compiled, not run on hardware
//
#include <hip/hip_runtime.h>
#include <hip/hip_bf16.h>
#include <cstdint>

// ---------------------------------------------------------------------------
// Types for WMMA fragments (wave32, gfx1250)
// ---------------------------------------------------------------------------
typedef __attribute__((ext_vector_type(16))) _Float16 v16h;
typedef __attribute__((ext_vector_type(8)))  _Float16 h8;
typedef __attribute__((ext_vector_type(8)))  float    v8f;

__device__ __forceinline__ float gelu_exact(float v) {
    return 0.5f * v * (1.0f + erff(v * 0.70710678118654752f));
}

// ---------------------------------------------------------------------------
// WMMA GEMM:  Y[M,N] = post( concat(X1,X2)[M,K] @ W[K,N] + bias )
// Block tile 128(M) x 32(N), K stepped by 32. 4 waves; wave w owns rows
// [32w, 32w+32) as two A fragments x two B fragments -> 4 accumulators,
// 4 x v_wmma_f32_16x16x32_f16 per wave per K-step.
// Register-prefetch + double-buffered LDS: tile k+1 is fetched into VGPRs
// while tile k feeds the WMMAs; a single s_barrier per K-step is sufficient
// (a wave can run at most one store-phase ahead, which targets the other
// LDS buffer).
// Fast staging path uses float2 (global_load_b64) vector loads (every K/N in
// this net is even -> 8B-aligned rows); guarded scalar path only on tile
// edges / the concat-K seam.
// ---------------------------------------------------------------------------
#define BM 128
#define BN 32
#define BK 32
#define LPAD 8

struct Seg16 { h8 lo, hi; };

__device__ __forceinline__ Seg16 fetch_a16(const float* __restrict__ X1,
                                           const float* __restrict__ X2,
                                           int K1, int K2, int K, int M,
                                           int m, int k0)
{
    Seg16 r;
    const bool in1 = (m < M) && (k0 + 16 <= K1);
    const bool in2 = (m < M) && X2 && (k0 >= K1) && (k0 + 16 <= K);
    if (in1 || in2) {
        const float* src = in1 ? (X1 + (size_t)m * K1 + k0)
                               : (X2 + (size_t)m * K2 + (k0 - K1));
        const float2* p = (const float2*)src;   // 8B-aligned (K even)
        #pragma unroll
        for (int i = 0; i < 4; ++i) {
            const float2 v = p[i];
            r.lo[2*i]   = (_Float16)v.x;
            r.lo[2*i+1] = (_Float16)v.y;
        }
        #pragma unroll
        for (int i = 0; i < 4; ++i) {
            const float2 v = p[4 + i];
            r.hi[2*i]   = (_Float16)v.x;
            r.hi[2*i+1] = (_Float16)v.y;
        }
    } else {
        #pragma unroll
        for (int i = 0; i < 16; ++i) {
            const int k = k0 + i;
            float v = 0.f;
            if (m < M && k < K) {
                v = (k < K1) ? X1[(size_t)m * K1 + k]
                             : X2[(size_t)m * K2 + (k - K1)];
            }
            if (i < 8) r.lo[i] = (_Float16)v; else r.hi[i - 8] = (_Float16)v;
        }
    }
    return r;
}

__device__ __forceinline__ h8 fetch_b8(const float* __restrict__ W,
                                       int K, int N, int k, int n0)
{
    h8 r;
    if (k < K && n0 + 8 <= N) {
        const float2* p = (const float2*)(W + (size_t)k * N + n0);
        #pragma unroll
        for (int i = 0; i < 4; ++i) {
            const float2 v = p[i];
            r[2*i]   = (_Float16)v.x;
            r[2*i+1] = (_Float16)v.y;
        }
    } else {
        #pragma unroll
        for (int i = 0; i < 8; ++i) {
            const int n = n0 + i;
            r[i] = (_Float16)((k < K && n < N) ? W[(size_t)k * N + n] : 0.f);
        }
    }
    return r;
}

__device__ __forceinline__ v16h frag_from_lds(const _Float16* p0,
                                              const _Float16* p1)
{
    const h8 lo = *(const h8*)p0;
    const h8 hi = *(const h8*)p1;
    return __builtin_shufflevector(lo, hi,
        0,1,2,3,4,5,6,7,8,9,10,11,12,13,14,15);
}

__global__ __launch_bounds__(128)
void wmma_gemm_kernel(const float* __restrict__ X1, const float* __restrict__ X2,
                      int K1, int K,
                      const float* __restrict__ W, const float* __restrict__ bias,
                      const float* __restrict__ residual,
                      const float* __restrict__ scale_log, int scale_idx,
                      float* __restrict__ Y, int M, int N, int act)
{
    __shared__ _Float16 As[2][BM][BK + LPAD];   // As[buf][m][k]
    __shared__ _Float16 Bs[2][BN][BK + LPAD];   // Bs[buf][n][k] (transposed W)

    const int tid  = threadIdx.x;
    const int wave = tid >> 5;
    const int lane = tid & 31;
    const int half = lane >> 4;     // 0: lanes 0-15, 1: lanes 16-31
    const int l16  = lane & 15;
    const int bm   = blockIdx.x * BM;
    const int bn   = blockIdx.y * BN;
    const int K2   = K - K1;

    // A staging: thread -> rows (tid>>1) and (tid>>1)+64, 16-wide k segment
    const int ar0 = tid >> 1;
    const int ar1 = ar0 + 64;
    const int aks = (tid & 1) * 16;
    // B staging: thread -> k (0..31), 8-wide n segment
    const int bk  = tid >> 2;
    const int bns = (tid & 3) * 8;

    v8f c00 = {0.f,0.f,0.f,0.f,0.f,0.f,0.f,0.f};
    v8f c01 = c00, c10 = c00, c11 = c00;

    // prefetch tile 0 into registers
    Seg16 pa0 = fetch_a16(X1, X2, K1, K2, K, M, bm + ar0, aks);
    Seg16 pa1 = fetch_a16(X1, X2, K1, K2, K, M, bm + ar1, aks);
    h8    pb  = fetch_b8(W, K, N, bk, bn + bns);

    int buf = 0;
    for (int kt = 0; kt < K; kt += BK) {
        // ---- commit prefetched tile to LDS[buf] ----
        { h8* d = (h8*)&As[buf][ar0][aks]; d[0] = pa0.lo; d[1] = pa0.hi; }
        { h8* d = (h8*)&As[buf][ar1][aks]; d[0] = pa1.lo; d[1] = pa1.hi; }
        #pragma unroll
        for (int i = 0; i < 8; ++i) Bs[buf][bns + i][bk] = pb[i];
        __syncthreads();

        // ---- prefetch tile k+1 (overlaps the WMMAs below) ----
        const int ktn = kt + BK;
        if (ktn < K) {
            pa0 = fetch_a16(X1, X2, K1, K2, K, M, bm + ar0, ktn + aks);
            pa1 = fetch_a16(X1, X2, K1, K2, K, M, bm + ar1, ktn + aks);
            pb  = fetch_b8(W, K, N, ktn + bk, bn + bns);
        }

        // ---- compute from LDS[buf] ----
        // A frags: lane<16 K{0..7,16..23}, lane>=16 K{8..15,24..31}
        const int r0 = wave * 32 + l16;
        const int r1 = r0 + 16;
        const v16h a0 = frag_from_lds(&As[buf][r0][half * 8],
                                      &As[buf][r0][16 + half * 8]);
        const v16h a1 = frag_from_lds(&As[buf][r1][half * 8],
                                      &As[buf][r1][16 + half * 8]);
        // B frags: lane<16 K{0..7,16..23}, lane>=16 K{8..15,24..31}
        const v16h b0 = frag_from_lds(&Bs[buf][l16][half * 8],
                                      &Bs[buf][l16][16 + half * 8]);
        const v16h b1 = frag_from_lds(&Bs[buf][16 + l16][half * 8],
                                      &Bs[buf][16 + l16][16 + half * 8]);

        c00 = __builtin_amdgcn_wmma_f32_16x16x32_f16(
            false, a0, false, b0, (short)0, c00, false, false);
        c01 = __builtin_amdgcn_wmma_f32_16x16x32_f16(
            false, a0, false, b1, (short)0, c01, false, false);
        c10 = __builtin_amdgcn_wmma_f32_16x16x32_f16(
            false, a1, false, b0, (short)0, c10, false, false);
        c11 = __builtin_amdgcn_wmma_f32_16x16x32_f16(
            false, a1, false, b1, (short)0, c11, false, false);

        buf ^= 1;
        // no second barrier needed: the next store phase targets the other
        // LDS buffer, and a wave cannot get two store-phases ahead.
    }

    const float smul = scale_log ? __expf(scale_log[scale_idx]) : 1.0f;

    // C/D layout: VGPR i -> M = i (+8 for lanes>=16), N = lane%16
    auto emit = [&](const v8f& a, int mbase, int n) {
        if (n >= N) return;
        #pragma unroll
        for (int i = 0; i < 8; ++i) {
            const int m = mbase + half * 8 + i;
            if (m < M) {
                float v = a[i];
                if (bias) v += bias[n];
                if (act == 1) v = gelu_exact(v);
                v *= smul;
                if (residual) v += residual[(size_t)m * N + n];
                Y[(size_t)m * N + n] = v;
            }
        }
    };
    emit(c00, bm + wave * 32,      bn + l16);
    emit(c01, bm + wave * 32,      bn + 16 + l16);
    emit(c10, bm + wave * 32 + 16, bn + l16);
    emit(c11, bm + wave * 32 + 16, bn + 16 + l16);
}

// ---------------------------------------------------------------------------
// LayerNorm: one wave32 per token (C <= 128), shfl_xor reductions.
// ---------------------------------------------------------------------------
__global__ __launch_bounds__(256)
void ln_kernel(const float* __restrict__ x, const float* __restrict__ g,
               const float* __restrict__ b, float* __restrict__ y, int M, int C)
{
    const int wave = threadIdx.x >> 5;
    const int lane = threadIdx.x & 31;
    const int t = blockIdx.x * 8 + wave;
    if (t >= M) return;
    const float* xr = x + (size_t)t * C;
    float vals[4];
    float sum = 0.f;
    const int nchunk = (C + 31) >> 5;
    #pragma unroll 4
    for (int i = 0; i < nchunk; ++i) {
        const int d = lane + i * 32;
        const float v = (d < C) ? xr[d] : 0.f;
        vals[i] = v; sum += v;
    }
    #pragma unroll
    for (int off = 16; off >= 1; off >>= 1) sum += __shfl_xor(sum, off, 32);
    const float mean = sum / (float)C;
    float vs = 0.f;
    #pragma unroll 4
    for (int i = 0; i < nchunk; ++i) {
        const int d = lane + i * 32;
        const float dv = (d < C) ? (vals[i] - mean) : 0.f;
        vs += dv * dv;
    }
    #pragma unroll
    for (int off = 16; off >= 1; off >>= 1) vs += __shfl_xor(vs, off, 32);
    const float inv = rsqrtf(vs / (float)C + 1e-5f);
    float* yr = y + (size_t)t * C;
    #pragma unroll 4
    for (int i = 0; i < nchunk; ++i) {
        const int d = lane + i * 32;
        if (d < C) yr[d] = (vals[i] - mean) * inv * g[d] + b[d];
    }
}

// ---------------------------------------------------------------------------
// Window attention core (LEAF=32) with RoPE, templated on head dim so the
// query row and score loops stay fully in VGPRs (hd in {8,13,20,32,82}).
// One wave per (window, head); lane = query row = K/V row staged to LDS.
// RoPE of K is folded into the LDS store (no kr register array).
// ---------------------------------------------------------------------------
template<int HD>
__global__ __launch_bounds__(64)
void attn_kernel_t(const float* __restrict__ qkv, float* __restrict__ o,
                   int numWin, int C, int H)
{
    constexpr int RD  = 2 * (HD / 2);
    constexpr int RH  = RD / 2;
    constexpr int STR = HD + ((HD & 1) ? 0 : 1);
    __shared__ float Ksm[2][32][STR];
    __shared__ float Vsm[2][32][STR];

    const int wslot = threadIdx.x >> 5;
    const int lane  = threadIdx.x & 31;
    const int p = blockIdx.x * 2 + wslot;
    if (p >= numWin * H) return;
    const int win  = p / H;
    const int head = p - win * H;
    const int t    = win * 32 + lane;
    const size_t base = (size_t)t * (3 * C) + head * HD;

    const float pos = (float)lane;
    float q[HD];
    #pragma unroll
    for (int d = 0; d < HD; ++d) q[d] = qkv[base + d];

    #pragma unroll
    for (int j = 0; j < RH; ++j) {
        const float ang = pos * __powf(10000.f, -2.f * (float)j / (float)RD);
        float sn, cs;
        __sincosf(ang, &sn, &cs);
        const float q1 = q[j], q2 = q[j + RH];
        q[j]      = q1 * cs - q2 * sn;
        q[j + RH] = q2 * cs + q1 * sn;
        const float k1 = qkv[base + C + j];
        const float k2 = qkv[base + C + j + RH];
        Ksm[wslot][lane][j]      = k1 * cs - k2 * sn;
        Ksm[wslot][lane][j + RH] = k2 * cs + k1 * sn;
    }
    if (HD & 1) Ksm[wslot][lane][HD - 1] = qkv[base + C + HD - 1];
    #pragma unroll
    for (int d = 0; d < HD; ++d) Vsm[wslot][lane][d] = qkv[base + 2 * C + d];
    // wave-synchronous: per-wave LDS ops are in order, no barrier needed.

    float s[32];
    const float scl = rsqrtf((float)HD);
    float mx = -1e30f;
    #pragma unroll 4
    for (int kk = 0; kk < 32; ++kk) {
        float a = 0.f;
        #pragma unroll
        for (int d = 0; d < HD; ++d) a += q[d] * Ksm[wslot][kk][d];
        a *= scl;
        s[kk] = a;
        mx = fmaxf(mx, a);
    }
    float denom = 0.f;
    #pragma unroll
    for (int kk = 0; kk < 32; ++kk) { s[kk] = __expf(s[kk] - mx); denom += s[kk]; }
    const float invd = 1.f / denom;
    #pragma unroll 2
    for (int d = 0; d < HD; ++d) {
        float a = 0.f;
        #pragma unroll
        for (int kk = 0; kk < 32; ++kk) a += s[kk] * Vsm[wslot][kk][d];
        o[(size_t)t * C + head * HD + d] = a * invd;
    }
}

// Dynamic fallback (unexpected head dims).
#define MAXHD 96
__global__ __launch_bounds__(64)
void attn_kernel_dyn(const float* __restrict__ qkv, float* __restrict__ o,
                     int numWin, int C, int H, int hd)
{
    __shared__ float Ksm[2][32][MAXHD];
    __shared__ float Vsm[2][32][MAXHD];
    const int wslot = threadIdx.x >> 5;
    const int lane  = threadIdx.x & 31;
    const int p = blockIdx.x * 2 + wslot;
    if (p >= numWin * H) return;
    const int win  = p / H;
    const int head = p - win * H;
    const int t    = win * 32 + lane;
    const int rd   = 2 * (hd / 2);
    const int rh   = rd >> 1;
    const size_t base = (size_t)t * (3 * C) + head * hd;

    float q[MAXHD];
    for (int d = 0; d < hd; ++d) {
        q[d] = qkv[base + d];
        Vsm[wslot][lane][d] = qkv[base + 2 * C + d];
    }
    const float pos = (float)lane;
    for (int j = 0; j < rh; ++j) {
        const float ang = pos * __powf(10000.f, -2.f * (float)j / (float)rd);
        float sn, cs;
        __sincosf(ang, &sn, &cs);
        const float q1 = q[j], q2 = q[j + rh];
        q[j]      = q1 * cs - q2 * sn;
        q[j + rh] = q2 * cs + q1 * sn;
        const float k1 = qkv[base + C + j];
        const float k2 = qkv[base + C + j + rh];
        Ksm[wslot][lane][j]      = k1 * cs - k2 * sn;
        Ksm[wslot][lane][j + rh] = k2 * cs + k1 * sn;
    }
    if (rd < hd) Ksm[wslot][lane][hd - 1] = qkv[base + C + hd - 1];

    float s[32];
    const float scl = rsqrtf((float)hd);
    float mx = -1e30f;
    for (int kk = 0; kk < 32; ++kk) {
        float a = 0.f;
        for (int d = 0; d < hd; ++d) a += q[d] * Ksm[wslot][kk][d];
        a *= scl;
        s[kk] = a;
        mx = fmaxf(mx, a);
    }
    float denom = 0.f;
    for (int kk = 0; kk < 32; ++kk) { s[kk] = __expf(s[kk] - mx); denom += s[kk]; }
    const float invd = 1.f / denom;
    for (int d = 0; d < hd; ++d) {
        float a = 0.f;
        for (int kk = 0; kk < 32; ++kk) a += s[kk] * Vsm[wslot][kk][d];
        o[(size_t)t * C + head * hd + d] = a * invd;
    }
}

// ---------------------------------------------------------------------------
// Embedding: h = center([pos,diag]) + sum_slots act * neighbor(feat).
// One wave per token; each lane owns 4 of 128 output channels.
// ---------------------------------------------------------------------------
#define IN_FEAT 124

__global__ __launch_bounds__(256)
void embed_kernel(const float* __restrict__ x,
                  const float* __restrict__ wc, const float* __restrict__ bc,
                  const float* __restrict__ wn, const float* __restrict__ bn,
                  float* __restrict__ out, int M)
{
    const int wave = threadIdx.x >> 5;
    const int lane = threadIdx.x & 31;
    const int t = blockIdx.x * 8 + wave;
    if (t >= M) return;
    const float* xr = x + (size_t)t * IN_FEAT;
    float h[4];
    #pragma unroll
    for (int c = 0; c < 4; ++c) {
        const int oo = lane + c * 32;
        float a = bc[oo];
        #pragma unroll
        for (int f = 0; f < 4; ++f) a += xr[f] * wc[f * 128 + oo];
        h[c] = a;
    }
    const float px = xr[0], py = xr[1], pz = xr[2];
    for (int s = 0; s < 24; ++s) {
        const float* nbp = xr + 4 + s * 5;
        const float w = nbp[4];
        if (w != 0.f) {   // uniform across the wave (same token)
            const float f0 = nbp[0];
            const float f1 = nbp[1] - px;
            const float f2 = nbp[2] - py;
            const float f3 = nbp[3] - pz;
            #pragma unroll
            for (int c = 0; c < 4; ++c) {
                const int oo = lane + c * 32;
                h[c] += bn[oo] + f0 * wn[0 * 128 + oo] + f1 * wn[1 * 128 + oo]
                              + f2 * wn[2 * 128 + oo] + f3 * wn[3 * 128 + oo]
                              + w  * wn[4 * 128 + oo];
            }
        }
    }
    float* orow = out + (size_t)t * 128;
    #pragma unroll
    for (int c = 0; c < 4; ++c) orow[lane + c * 32] = h[c];
}

// ---------------------------------------------------------------------------
// Host orchestration
// ---------------------------------------------------------------------------
namespace {
struct Lin { const float* b; const float* w; };
struct TB  { Lin fc1, fc2, proj, qkv;
             const float *ln1b, *ln1g, *ln2b, *ln2g; };

constexpr int RANKS[5] = {128, 82, 52, 32, 20};
constexpr int ENC[5]   = {20, 32, 52, 82, 128};
constexpr int DECD[5]  = {128, 82, 52, 32, 20};
constexpr int M0 = 4 * 32768;                 // B * N tokens
constexpr size_t ACT_CAP = (size_t)M0 * 20;   // max [M,C] after enc_in
constexpr size_t SCR_CAP = (size_t)M0 * 80;   // max of qkv (M*3C) and mlp (M*4C)
}

extern "C" void kernel_launch(void* const* d_in, const int* in_sizes, int n_in,
                              void* d_out, int out_size, void* d_ws, size_t ws_size,
                              hipStream_t stream) {
    (void)in_sizes; (void)n_in; (void)out_size; (void)ws_size;

    // ---- gather params: x first, then jax-pytree (sorted-key) flatten ----
    int idx = 0;
    const float* Xin = (const float*)d_in[idx++];
    auto nextp   = [&]() { return (const float*)d_in[idx++]; };
    auto nextLin = [&]() { Lin l; l.b = nextp(); l.w = nextp(); return l; };
    auto nextTB  = [&]() {
        TB t;
        t.fc1 = nextLin(); t.fc2 = nextLin();
        t.ln1b = nextp(); t.ln1g = nextp();
        t.ln2b = nextp(); t.ln2g = nextp();
        t.proj = nextLin(); t.qkv = nextLin();
        return t;
    };

    TB  bottleneck = nextTB();
    Lin center     = nextLin();
    TB  dec_tb[5];  for (int i = 0; i < 5; ++i) dec_tb[i] = nextTB();
    Lin down[5];    for (int i = 0; i < 5; ++i) down[i]   = nextLin();
    TB  enc_tb[5];  for (int i = 0; i < 5; ++i) enc_tb[i] = nextTB();
    Lin enc_in     = nextLin();
    Lin heads_u[5], heads_v[5];
    for (int i = 0; i < 5; ++i) { heads_u[i] = nextLin(); heads_v[i] = nextLin(); }
    Lin leaf_head  = nextLin();
    Lin leaf_proj  = nextLin();
    const float* log_scale_leaf = nextp();
    const float* log_scales     = nextp();
    Lin neighbor   = nextLin();
    Lin skip_fuse[5]; for (int i = 0; i < 5; ++i) skip_fuse[i] = nextLin();
    Lin skip_proj[5]; for (int i = 0; i < 5; ++i) skip_proj[i] = nextLin();
    Lin up[5];        for (int i = 0; i < 5; ++i) up[i]        = nextLin();

    // ---- arena allocator over d_ws ----
    char* ws = (char*)d_ws;
    size_t wsoff = 0;
    auto alloc = [&](size_t nfloats) -> float* {
        float* p = (float*)(ws + wsoff);
        wsoff += ((nfloats * sizeof(float) + 255) / 256) * 256;
        return p;
    };
    float* embuf  = alloc((size_t)M0 * 128);  // embed output [M0,128]
    float* bufA   = alloc(ACT_CAP);
    float* bufB   = alloc(ACT_CAP);
    float* lnbuf  = alloc(ACT_CAP);           // LN output, also attn-o
    float* scratch= alloc(SCR_CAP);           // qkv then MLP hidden
    float* skproj = alloc(ACT_CAP);
    float* skip[5];
    for (int j = 0; j < 5; ++j) skip[j] = alloc((size_t)(M0 >> j) * ENC[j]);

    auto gemm = [&](const float* X1, const float* X2, int K1, int K,
                    const Lin& lin, const float* resid,
                    const float* slog, int sidx,
                    float* Y, int M, int N, int act) {
        dim3 grid((M + BM - 1) / BM, (N + BN - 1) / BN);
        wmma_gemm_kernel<<<grid, dim3(128), 0, stream>>>(
            X1, X2, K1, K, lin.w, lin.b, resid, slog, sidx, Y, M, N, act);
    };
    auto layernorm = [&](const float* x, const float* g, const float* b,
                         float* y, int M, int C) {
        ln_kernel<<<dim3((M + 7) / 8), dim3(256), 0, stream>>>(x, g, b, y, M, C);
    };
    auto launch_attn = [&](const float* qkvp, float* op,
                           int numWin, int C, int H, int hd) {
        const dim3 grid((numWin * H + 1) / 2), blk(64);
        switch (hd) {
        case 8:  attn_kernel_t<8 ><<<grid, blk, 0, stream>>>(qkvp, op, numWin, C, H); break;
        case 13: attn_kernel_t<13><<<grid, blk, 0, stream>>>(qkvp, op, numWin, C, H); break;
        case 20: attn_kernel_t<20><<<grid, blk, 0, stream>>>(qkvp, op, numWin, C, H); break;
        case 32: attn_kernel_t<32><<<grid, blk, 0, stream>>>(qkvp, op, numWin, C, H); break;
        case 82: attn_kernel_t<82><<<grid, blk, 0, stream>>>(qkvp, op, numWin, C, H); break;
        default: attn_kernel_dyn  <<<grid, blk, 0, stream>>>(qkvp, op, numWin, C, H, hd); break;
        }
    };
    auto run_tblock = [&](const TB& tb, float* xbuf, float* ybuf, int M, int C) {
        const int H  = (C % 4 == 0) ? 4 : 1;
        const int hd = C / H;
        layernorm(xbuf, tb.ln1g, tb.ln1b, lnbuf, M, C);
        gemm(lnbuf, nullptr, C, C, tb.qkv, nullptr, nullptr, 0, scratch, M, 3 * C, 0);
        launch_attn(scratch, lnbuf, M / 32, C, H, hd);
        gemm(lnbuf, nullptr, C, C, tb.proj, xbuf, nullptr, 0, ybuf, M, C, 0);
        layernorm(ybuf, tb.ln2g, tb.ln2b, lnbuf, M, C);
        gemm(lnbuf, nullptr, C, C, tb.fc1, nullptr, nullptr, 0, scratch, M, 4 * C, 1);
        gemm(scratch, nullptr, 4 * C, 4 * C, tb.fc2, ybuf, nullptr, 0, xbuf, M, C, 0);
    };

    // ---- embed + enc_in ----
    embed_kernel<<<dim3((M0 + 7) / 8), dim3(256), 0, stream>>>(
        Xin, center.w, center.b, neighbor.w, neighbor.b, embuf, M0);
    float* cur = bufA;
    float* alt = bufB;
    gemm(embuf, nullptr, 128, 128, enc_in, nullptr, nullptr, 0, cur, M0, ENC[0], 0);

    // ---- encoder ----
    int M = M0;
    for (int j = 0; j < 5; ++j) {
        run_tblock(enc_tb[j], cur, alt, M, ENC[j]);
        hipMemcpyAsync(skip[j], cur, (size_t)M * ENC[j] * sizeof(float),
                       hipMemcpyDeviceToDevice, stream);
        const int Cn = (j + 1 < 5) ? ENC[j + 1] : ENC[4];
        // [M,C] reinterpreted as [M/2, 2C] (layout no-op)
        gemm(cur, nullptr, 2 * ENC[j], 2 * ENC[j], down[j],
             nullptr, nullptr, 0, alt, M / 2, Cn, 0);
        { float* t = cur; cur = alt; alt = t; }
        M /= 2;
    }

    // ---- bottleneck ----
    run_tblock(bottleneck, cur, alt, M, ENC[4]);

    // ---- decoder ----
    int prevC = ENC[4];
    for (int i = 0; i < 5; ++i) {
        const int d = DECD[i];
        gemm(cur, nullptr, prevC, prevC, up[i], nullptr, nullptr, 0,
             alt, M, 2 * d, 0);                       // then view [2M, d]
        { float* t = cur; cur = alt; alt = t; }
        M *= 2;
        gemm(skip[4 - i], nullptr, ENC[4 - i], ENC[4 - i], skip_proj[i],
             nullptr, nullptr, 0, skproj, M, d, 0);
        gemm(cur, skproj, d, 2 * d, skip_fuse[i],     // concat-K fuse
             nullptr, nullptr, 0, alt, M, d, 0);
        { float* t = cur; cur = alt; alt = t; }
        run_tblock(dec_tb[i], cur, alt, M, d);
        prevC = d;
    }

    // ---- heads: (leaf, u0,v0, ..., u4,v4) concatenated in d_out ----
    float* out = (float*)d_out;
    size_t off = (size_t)M0 * 32;  // leaf goes first
    for (int i = 0; i < 5; ++i) {
        gemm(cur, nullptr, 20, 20, heads_u[i], nullptr, log_scales, i,
             out + off, M0, RANKS[i], 0);
        off += (size_t)M0 * RANKS[i];
        gemm(cur, nullptr, 20, 20, heads_v[i], nullptr, log_scales, i,
             out + off, M0, RANKS[i], 0);
        off += (size_t)M0 * RANKS[i];
    }
    gemm(cur, nullptr, 20, 20, leaf_proj, nullptr, nullptr, 0, lnbuf, M0, 32, 0);
    gemm(lnbuf, nullptr, 32, 32, leaf_head, nullptr, log_scale_leaf, 0,
         out, M0, 32, 0);
}
